// MultiScaleGraph_40114994545134
// MI455X (gfx1250) — matrix operand order repaired and checked by reading, not compiled, not run on hardware
//
#include <hip/hip_runtime.h>
#include <cstdint>
#include <cstddef>

namespace {

constexpr int kN0 = 100000, kN1 = 25000, kN2 = 6250;
constexpr int kF = 128, kOut = 5;

typedef _Float16 v8h  __attribute__((ext_vector_type(8)));
typedef _Float16 v16h __attribute__((ext_vector_type(16)));
typedef float    v8f  __attribute__((ext_vector_type(8)));

// ================= WMMA GEMM: C[N,128] = (A[N,Kin] .* row_scale) @ W[Kin,128] + bias =================
// 256 threads = 8 waves; each wave computes a 16x128 output slab via 8 accumulators of
// v_wmma_f32_16x16x32_f16. Kin must be a multiple of 32 (here 32 or 128).
__global__ __launch_bounds__(256)
void gemm_wmma_kernel(const float* __restrict__ A, const float* __restrict__ W,
                      const float* __restrict__ bias, const float* __restrict__ row_scale,
                      float* __restrict__ C, int N, int Kin)
{
    constexpr int SW = 136;                 // padded LDS row stride in halves (272B, 16B-aligned)
    __shared__ _Float16 Wt[128 * SW];       // W transposed to [n][k], f16
    __shared__ _Float16 Als[128 * SW];      // 8 waves x 16 rows of A, f16

    const int tid  = threadIdx.x;
    const int wave = tid >> 5;
    const int lane = tid & 31;
    const int l    = lane & 15;
    const int hi   = lane >> 4;
    const int rowBase = blockIdx.x * 128 + wave * 16;

    // Stage W^T into LDS as f16 (coalesced global reads)
    for (int idx = tid; idx < Kin * 128; idx += 256) {
        int k = idx >> 7;                  // W row
        int n = idx & 127;                 // W col
        Wt[n * SW + k] = (_Float16)W[idx];
    }
    // Stage this wave's 16 A rows into LDS as f16 (with optional row scaling)
    for (int idx = lane; idx < 16 * Kin; idx += 32) {
        int r = idx / Kin;
        int c = idx - r * Kin;
        int grow = rowBase + r;
        float v = 0.0f;
        if (grow < N) {
            v = A[(size_t)grow * Kin + c];
            if (row_scale) v *= row_scale[grow];
        }
        Als[(wave * 16 + r) * SW + c] = (_Float16)v;
    }
    __syncthreads();

    v8f acc[8] = {};
    const _Float16* Arow = &Als[(wave * 16 + l) * SW];
    const int ksteps = Kin >> 5;
    for (int kk = 0; kk < ksteps; ++kk) {
        const int k0 = kk * 32;
        // A fragment per ISA 16-bit A layout: j<8 -> K=j+8*hi ; j>=8 -> K=8+j+8*hi
        v8h a0 = *(const v8h*)(Arow + k0 + 8 * hi);
        v8h a1 = *(const v8h*)(Arow + k0 + 16 + 8 * hi);
        v16h a = __builtin_shufflevector(a0, a1, 0,1,2,3,4,5,6,7,8,9,10,11,12,13,14,15);
        #pragma unroll
        for (int t = 0; t < 8; ++t) {
            // B fragment: j<8 -> K=j+8*hi ; j>=8 -> K=8+j+8*hi  (read from transposed LDS copy)
            const _Float16* Wrow = &Wt[(t * 16 + l) * SW + k0 + 8 * hi];
            v8h b0 = *(const v8h*)(Wrow);
            v8h b1 = *(const v8h*)(Wrow + 16);
            v16h b = __builtin_shufflevector(b0, b1, 0,1,2,3,4,5,6,7,8,9,10,11,12,13,14,15);
            acc[t] = __builtin_amdgcn_wmma_f32_16x16x32_f16(
                false, a, false, b, (short)0, acc[t], false, false);
        }
    }

    // C/D layout: VGPR r holds M=r (lanes 0-15) and M=8+r (lanes 16-31), N=lane&15
    #pragma unroll
    for (int t = 0; t < 8; ++t) {
        const int col = t * 16 + l;
        const float bv = bias[col];
        #pragma unroll
        for (int r = 0; r < 8; ++r) {
            const int row = rowBase + r + 8 * hi;
            if (row < N) C[(size_t)row * 128 + col] = acc[t][r] + bv;
        }
    }
}

// ================= edge-degree / count kernels =================
__global__ void edge_deg_kernel(const int* __restrict__ src, const int* __restrict__ dst,
                                int* __restrict__ outdeg, int* __restrict__ indeg, int E)
{
    int i = blockIdx.x * blockDim.x + threadIdx.x;
    if (i < E) {
        atomicAdd(&outdeg[src[i]], 1);
        atomicAdd(&indeg[dst[i]], 1);
    }
}

__global__ void dst_count_kernel(const int* __restrict__ dst, int* __restrict__ cnt, int E)
{
    int i = blockIdx.x * blockDim.x + threadIdx.x;
    if (i < E) atomicAdd(&cnt[dst[i]], 1);
}

__global__ void invsqrt_deg_kernel(const int* __restrict__ deg, float* __restrict__ out, int n)
{
    int i = blockIdx.x * blockDim.x + threadIdx.x;
    if (i < n) {
        int d = deg[i];
        out[i] = (d > 0) ? rsqrtf((float)d) : 0.0f;
    }
}

// ================= wave-per-edge scatter-add of 128-float rows =================
__global__ void scatter_add_kernel(const float* __restrict__ feat, const int* __restrict__ src,
                                   const int* __restrict__ dst, const float* __restrict__ scale,
                                   float* __restrict__ out, int E)
{
    int e = (int)((blockIdx.x * blockDim.x + threadIdx.x) >> 5);
    int lane = threadIdx.x & 31;
    if (e >= E) return;
    int s = src[e], d = dst[e];
    float sc = scale ? scale[s] : 1.0f;
    float4 v = ((const float4*)(feat + (size_t)s * kF))[lane];
    float* o = out + (size_t)d * kF + lane * 4;
    __hip_atomic_fetch_add(&o[0], v.x * sc, __ATOMIC_RELAXED, __HIP_MEMORY_SCOPE_AGENT);
    __hip_atomic_fetch_add(&o[1], v.y * sc, __ATOMIC_RELAXED, __HIP_MEMORY_SCOPE_AGENT);
    __hip_atomic_fetch_add(&o[2], v.z * sc, __ATOMIC_RELAXED, __HIP_MEMORY_SCOPE_AGENT);
    __hip_atomic_fetch_add(&o[3], v.w * sc, __ATOMIC_RELAXED, __HIP_MEMORY_SCOPE_AGENT);
}

// ================= in-place mean divide: s[n][:] /= max(cnt[n],1) =================
__global__ void mean_div_kernel(float* __restrict__ s, const int* __restrict__ cnt, int n)
{
    int i = blockIdx.x * blockDim.x + threadIdx.x;
    if (i < n * 32) {
        int node = i >> 5;
        float c = fmaxf((float)cnt[node], 1.0f);
        float4 v = ((const float4*)s)[i];
        v.x /= c; v.y /= c; v.z /= c; v.w /= c;
        ((float4*)s)[i] = v;
    }
}

// ================= head: out = softmax(concat(g,d) @ W[256,5] + b), wave per node =================
__global__ __launch_bounds__(256)
void head_softmax_kernel(const float* __restrict__ g, const float* __restrict__ d,
                         const float* __restrict__ W, const float* __restrict__ b,
                         float* __restrict__ out, int N)
{
    __shared__ float Ws[2 * kF * kOut];
    __shared__ float bs[kOut];
    for (int i = threadIdx.x; i < 2 * kF * kOut; i += blockDim.x) Ws[i] = W[i];
    if (threadIdx.x < kOut) bs[threadIdx.x] = b[threadIdx.x];
    __syncthreads();

    int node = blockIdx.x * 8 + (threadIdx.x >> 5);
    int lane = threadIdx.x & 31;
    if (node >= N) return;

    float p[kOut] = {0.f, 0.f, 0.f, 0.f, 0.f};
    const float* gr = g + (size_t)node * kF;
    const float* dr = d + (size_t)node * kF;
    #pragma unroll
    for (int c = 0; c < 4; ++c) {
        int r = c * 32 + lane;
        float x = gr[r];
        #pragma unroll
        for (int j = 0; j < kOut; ++j) p[j] += x * Ws[r * kOut + j];
    }
    #pragma unroll
    for (int c = 0; c < 4; ++c) {
        int r = c * 32 + lane;
        float x = dr[r];
        #pragma unroll
        for (int j = 0; j < kOut; ++j) p[j] += x * Ws[(kF + r) * kOut + j];
    }
    #pragma unroll
    for (int off = 16; off > 0; off >>= 1) {
        #pragma unroll
        for (int j = 0; j < kOut; ++j) p[j] += __shfl_down(p[j], off, 32);
    }
    if (lane == 0) {
        float m = -3.0e38f;
        #pragma unroll
        for (int j = 0; j < kOut; ++j) { p[j] += bs[j]; m = fmaxf(m, p[j]); }
        float sum = 0.f;
        #pragma unroll
        for (int j = 0; j < kOut; ++j) { p[j] = __expf(p[j] - m); sum += p[j]; }
        float inv = 1.0f / sum;
        #pragma unroll
        for (int j = 0; j < kOut; ++j) out[(size_t)node * kOut + j] = p[j] * inv;
    }
}

inline void launch_gemm(const float* A, const float* W, const float* b, const float* rs,
                        float* C, int N, int Kin, hipStream_t s)
{
    gemm_wmma_kernel<<<dim3((N + 127) / 128), 256, 0, s>>>(A, W, b, rs, C, N, Kin);
}

} // namespace

extern "C" void kernel_launch(void* const* d_in, const int* in_sizes, int n_in,
                              void* d_out, int out_size, void* d_ws, size_t ws_size,
                              hipStream_t stream)
{
    // ---- inputs (setup_inputs order) ----
    const float* X        = (const float*)d_in[0];
    const int*   g0_src   = (const int*)d_in[1];   const int* g0_dst  = (const int*)d_in[2];
    const int*   g1_src   = (const int*)d_in[3];   const int* g1_dst  = (const int*)d_in[4];
    const int*   g2_src   = (const int*)d_in[5];   const int* g2_dst  = (const int*)d_in[6];
    const int*   inc0_src = (const int*)d_in[7];   const int* inc0_dst= (const int*)d_in[8];
    const int*   inc1_src = (const int*)d_in[9];   const int* inc1_dst= (const int*)d_in[10];
    const int*   dec0_src = (const int*)d_in[11];  const int* dec0_dst= (const int*)d_in[12];
    const int*   dec1_src = (const int*)d_in[13];  const int* dec1_dst= (const int*)d_in[14];
    const float* W_emb  = (const float*)d_in[15];  const float* b_emb  = (const float*)d_in[16];
    const float* W_gcn0 = (const float*)d_in[17];  const float* b_gcn0 = (const float*)d_in[18];
    const float* W_gcn1 = (const float*)d_in[19];  const float* b_gcn1 = (const float*)d_in[20];
    const float* W_gcn2 = (const float*)d_in[21];  const float* b_gcn2 = (const float*)d_in[22];
    const float* W_enc0 = (const float*)d_in[23];  const float* b_enc0 = (const float*)d_in[24];
    const float* W_enc1 = (const float*)d_in[25];  const float* b_enc1 = (const float*)d_in[26];
    const float* W_dec0 = (const float*)d_in[27];  const float* b_dec0 = (const float*)d_in[28];
    const float* W_dec1 = (const float*)d_in[29];  const float* b_dec1 = (const float*)d_in[30];
    const float* W_h0 = (const float*)d_in[31];    const float* b_h0 = (const float*)d_in[32];
    const float* W_h1 = (const float*)d_in[33];    const float* b_h1 = (const float*)d_in[34];
    const float* W_h2 = (const float*)d_in[35];    const float* b_h2 = (const float*)d_in[36];
    float* out = (float*)d_out;

    const int E_g0 = in_sizes[1], E_g1 = in_sizes[3], E_g2 = in_sizes[5];
    const int E_inc0 = in_sizes[7], E_inc1 = in_sizes[9];
    const int E_dec0 = in_sizes[11], E_dec1 = in_sizes[13];

    // ---- workspace carve ----
    const size_t N0F = (size_t)kN0 * kF, N1F = (size_t)kN1 * kF, N2F = (size_t)kN2 * kF;
    char* p = (char*)d_ws;
    auto allocf = [&](size_t n) { float* r = (float*)p; p += n * sizeof(float); return r; };
    float* h0  = allocf(N0F);   // h0, later reused as d0 accumulation
    float* bB  = allocf(N0F);   // e0 -> agg0 -> g0
    float* h1  = allocf(N1F);   // h1, later reused as d1
    float* bE  = allocf(N1F);   // e1 -> agg1 -> g1
    float* t2  = allocf(N1F);   // d1 @ W_dec1
    float* h2  = allocf(N2F);   // h2
    float* bG  = allocf(N2F);   // agg2 -> g2 (= d2)
    float* tb  = allocf(N2F);   // g2 @ W_dec0
    float* ns  = allocf(kN0);
    float* nd  = allocf(kN0);
    int* degO = (int*)allocf(kN0);
    int* degI = (int*)allocf(kN0);
    int* cnt  = (int*)allocf(kN0);

    auto eb = [](int E) { return dim3((unsigned)((E + 255) / 256)); };   // thread-per-edge
    auto ew = [](int E) { return dim3((unsigned)((E + 7) / 8)); };       // wave-per-edge
    auto nb = [](int n) { return dim3((unsigned)((n + 255) / 256)); };

    // 1) h0 = X @ W_emb + b_emb
    launch_gemm(X, W_emb, b_emb, nullptr, h0, kN0, 32, stream);
    // 2) e0 = h0 @ W_enc0 + b_enc0
    launch_gemm(h0, W_enc0, b_enc0, nullptr, bB, kN0, 128, stream);
    // 3) h1 = mean_agg(e0, inc0)
    hipMemsetAsync(h1, 0, N1F * sizeof(float), stream);
    hipMemsetAsync(cnt, 0, kN1 * sizeof(int), stream);
    dst_count_kernel<<<eb(E_inc0), 256, 0, stream>>>(inc0_dst, cnt, E_inc0);
    scatter_add_kernel<<<ew(E_inc0), 256, 0, stream>>>(bB, inc0_src, inc0_dst, nullptr, h1, E_inc0);
    mean_div_kernel<<<nb(kN1 * 32), 256, 0, stream>>>(h1, cnt, kN1);
    // 4) e1 = h1 @ W_enc1 + b_enc1
    launch_gemm(h1, W_enc1, b_enc1, nullptr, bE, kN1, 128, stream);
    // 5) h2 = mean_agg(e1, inc1)
    hipMemsetAsync(h2, 0, N2F * sizeof(float), stream);
    hipMemsetAsync(cnt, 0, kN2 * sizeof(int), stream);
    dst_count_kernel<<<eb(E_inc1), 256, 0, stream>>>(inc1_dst, cnt, E_inc1);
    scatter_add_kernel<<<ew(E_inc1), 256, 0, stream>>>(bE, inc1_src, inc1_dst, nullptr, h2, E_inc1);
    mean_div_kernel<<<nb(kN2 * 32), 256, 0, stream>>>(h2, cnt, kN2);

    // 6) GraphConv scale 0: g0 = (scatter(h0*ns) * nd) @ W_gcn0 + b   (in place in bB)
    hipMemsetAsync(degO, 0, kN0 * sizeof(int), stream);
    hipMemsetAsync(degI, 0, kN0 * sizeof(int), stream);
    edge_deg_kernel<<<eb(E_g0), 256, 0, stream>>>(g0_src, g0_dst, degO, degI, E_g0);
    invsqrt_deg_kernel<<<nb(kN0), 256, 0, stream>>>(degO, ns, kN0);
    invsqrt_deg_kernel<<<nb(kN0), 256, 0, stream>>>(degI, nd, kN0);
    hipMemsetAsync(bB, 0, N0F * sizeof(float), stream);
    scatter_add_kernel<<<ew(E_g0), 256, 0, stream>>>(h0, g0_src, g0_dst, ns, bB, E_g0);
    launch_gemm(bB, W_gcn0, b_gcn0, nd, bB, kN0, 128, stream);

    // 7) GraphConv scale 1 (in place in bE)
    hipMemsetAsync(degO, 0, kN1 * sizeof(int), stream);
    hipMemsetAsync(degI, 0, kN1 * sizeof(int), stream);
    edge_deg_kernel<<<eb(E_g1), 256, 0, stream>>>(g1_src, g1_dst, degO, degI, E_g1);
    invsqrt_deg_kernel<<<nb(kN1), 256, 0, stream>>>(degO, ns, kN1);
    invsqrt_deg_kernel<<<nb(kN1), 256, 0, stream>>>(degI, nd, kN1);
    hipMemsetAsync(bE, 0, N1F * sizeof(float), stream);
    scatter_add_kernel<<<ew(E_g1), 256, 0, stream>>>(h1, g1_src, g1_dst, ns, bE, E_g1);
    launch_gemm(bE, W_gcn1, b_gcn1, nd, bE, kN1, 128, stream);

    // 8) GraphConv scale 2 (into bG)
    hipMemsetAsync(degO, 0, kN2 * sizeof(int), stream);
    hipMemsetAsync(degI, 0, kN2 * sizeof(int), stream);
    edge_deg_kernel<<<eb(E_g2), 256, 0, stream>>>(g2_src, g2_dst, degO, degI, E_g2);
    invsqrt_deg_kernel<<<nb(kN2), 256, 0, stream>>>(degO, ns, kN2);
    invsqrt_deg_kernel<<<nb(kN2), 256, 0, stream>>>(degI, nd, kN2);
    hipMemsetAsync(bG, 0, N2F * sizeof(float), stream);
    scatter_add_kernel<<<ew(E_g2), 256, 0, stream>>>(h2, g2_src, g2_dst, ns, bG, E_g2);
    launch_gemm(bG, W_gcn2, b_gcn2, nd, bG, kN2, 128, stream);

    // 9) d1 = mean_agg(g2 @ W_dec0 + b_dec0, dec0)   (d1 reuses h1 buffer)
    launch_gemm(bG, W_dec0, b_dec0, nullptr, tb, kN2, 128, stream);
    hipMemsetAsync(h1, 0, N1F * sizeof(float), stream);
    hipMemsetAsync(cnt, 0, kN1 * sizeof(int), stream);
    dst_count_kernel<<<eb(E_dec0), 256, 0, stream>>>(dec0_dst, cnt, E_dec0);
    scatter_add_kernel<<<ew(E_dec0), 256, 0, stream>>>(tb, dec0_src, dec0_dst, nullptr, h1, E_dec0);
    mean_div_kernel<<<nb(kN1 * 32), 256, 0, stream>>>(h1, cnt, kN1);

    // 10) d0 = mean_agg(d1 @ W_dec1 + b_dec1, dec1)  (d0 reuses h0 buffer)
    launch_gemm(h1, W_dec1, b_dec1, nullptr, t2, kN1, 128, stream);
    hipMemsetAsync(h0, 0, N0F * sizeof(float), stream);
    hipMemsetAsync(cnt, 0, kN0 * sizeof(int), stream);
    dst_count_kernel<<<eb(E_dec1), 256, 0, stream>>>(dec1_dst, cnt, E_dec1);
    scatter_add_kernel<<<ew(E_dec1), 256, 0, stream>>>(t2, dec1_src, dec1_dst, nullptr, h0, E_dec1);
    mean_div_kernel<<<nb(kN0 * 32), 256, 0, stream>>>(h0, cnt, kN0);

    // 11) heads (wave per node); d2 == g2
    head_softmax_kernel<<<dim3((kN0 + 7) / 8), 256, 0, stream>>>(bB, h0, W_h0, b_h0, out, kN0);
    head_softmax_kernel<<<dim3((kN1 + 7) / 8), 256, 0, stream>>>(bE, h1, W_h1, b_h1,
                                                                 out + (size_t)kN0 * kOut, kN1);
    head_softmax_kernel<<<dim3((kN2 + 7) / 8), 256, 0, stream>>>(bG, bG, W_h2, b_h2,
                                                                 out + (size_t)(kN0 + kN1) * kOut, kN2);
}
